// HeteroNet_47416438948428
// MI455X (gfx1250) — matrix operand order repaired and checked by reading, not compiled, not run on hardware
//
#include <hip/hip_runtime.h>
#include <hip/hip_bf16.h>
#include <math.h>

#define HID   128
#define NLIG  50000
#define NPROT 150000
#define NEDGE 500000
#define NGRAPH 256

typedef __attribute__((ext_vector_type(16))) __bf16        v16bf;
typedef __attribute__((ext_vector_type(8)))  float         v8f;
typedef __attribute__((ext_vector_type(8)))  unsigned int  v8u;

// ---- bf16 helpers: use native f32->bf16 conversion hardware ----
static __device__ __forceinline__ unsigned short bf16b(float f) {
  return __builtin_bit_cast(unsigned short, (__bf16)f);
}
static __device__ __forceinline__ unsigned packbf(float lo, float hi) {
  return ((unsigned)bf16b(hi) << 16) | (unsigned)bf16b(lo);
}

// ---- WMMA wrapper: D = A(16x32 bf16) x B(32x16 bf16) + C(f32) ----
static __device__ __forceinline__ v8f wmma_bf16(v8u a, v8u b, v8f c) {
  return __builtin_amdgcn_wmma_f32_16x16x32_bf16(
      false, __builtin_bit_cast(v16bf, a),
      false, __builtin_bit_cast(v16bf, b),
      (short)0, c, false, false);
}

// B fragment: pre-swizzled so lane reads 8 consecutive dwords (2x b128)
static __device__ __forceinline__ v8u load_wfrag(const unsigned* __restrict__ base,
                                                 int frag, int lane) {
  const unsigned* p = base + ((size_t)frag * 32 + lane) * 8;
  v8u r;
#pragma unroll
  for (int v = 0; v < 8; ++v) r[v] = p[v];
  return r;
}

// A-matrix 16x32 bf16 layout: K offset held by (lane-half, vgpr)
static __device__ __forceinline__ int a_k0(int s, int v, int half) {
  return s * 32 + half * 16 + v * 2;
}

// ---------------------------------------------------------------------------
// Weight prep: fp32 [K][N] row-major -> bf16x2 swizzled B-fragment order
// dword index = ((n*Ksteps + s)*32 + lane)*8 + v ; content = (W[k0][nn], W[k0+1][nn])
// nn = n*16 + (lane&15), k0 = 32s + (lane>>4)*16 + 2v ; zero-pad k >= Kreal
// ---------------------------------------------------------------------------
__global__ void prep_weight(const float* __restrict__ W, unsigned* __restrict__ dst,
                            int Kreal, int Kpad, int N) {
  int idx = blockIdx.x * blockDim.x + threadIdx.x;
  int total = (Kpad >> 1) * N;
  if (idx >= total) return;
  int v = idx & 7;
  int t = idx >> 3;
  int lane = t & 31;
  int frag = t >> 5;
  int Ksteps = Kpad >> 5;
  int s = frag % Ksteps;
  int n = frag / Ksteps;
  int nn = n * 16 + (lane & 15);
  int k0 = s * 32 + (lane >> 4) * 16 + 2 * v;
  float lo = (k0     < Kreal) ? W[(size_t)k0 * N + nn]       : 0.0f;
  float hi = (k0 + 1 < Kreal) ? W[(size_t)(k0 + 1) * N + nn] : 0.0f;
  dst[idx] = packbf(lo, hi);
}

// ---------------------------------------------------------------------------
// Node embedding: out[N,128] = X[N,44] @ W_node (K padded 44 -> 64)
// ---------------------------------------------------------------------------
__global__ void node_embed(const float* __restrict__ X, const unsigned* __restrict__ Wswz,
                           float* __restrict__ out, int ntiles) {
  int lane = threadIdx.x & 31;
  int wave = threadIdx.x >> 5;
  int tile = blockIdx.x * 4 + wave;
  if (tile >= ntiles) return;
  int half = lane >> 4, M = lane & 15;
  const float* xr = X + (size_t)(tile * 16 + M) * 44;
  v8u a[2];
#pragma unroll
  for (int s = 0; s < 2; ++s)
#pragma unroll
    for (int v = 0; v < 8; ++v) {
      int k0 = a_k0(s, v, half);
      a[s][v] = (k0 < 44) ? packbf(xr[k0], xr[k0 + 1]) : 0u;
    }
#pragma unroll
  for (int n = 0; n < 8; ++n) {
    v8f c = {};
    c = wmma_bf16(a[0], load_wfrag(Wswz, n * 2 + 0, lane), c);
    c = wmma_bf16(a[1], load_wfrag(Wswz, n * 2 + 1, lane), c);
#pragma unroll
    for (int r = 0; r < 8; ++r)
      out[(size_t)(tile * 16 + r + 8 * half) * HID + n * 16 + M] = c[r];
  }
}

// ---------------------------------------------------------------------------
// Edge message: base = silu(rbf @ Wb + bb);  msg = (base * x_src[src]) @ Wn
// atomically accumulated into agg[dst], deg[dst].  One wave = 16 edges.
// ---------------------------------------------------------------------------
__global__ void edge_msg(const float* __restrict__ eattr,
                         const int* __restrict__ idx_src, const int* __restrict__ idx_dst,
                         const float* __restrict__ x_src,
                         const unsigned* __restrict__ WbSwz, const float* __restrict__ bb,
                         const unsigned* __restrict__ WnSwz,
                         float* __restrict__ agg, float* __restrict__ deg, int ntiles) {
  __shared__ unsigned lds[4][16 * 64];  // per-wave 16x128 bf16 staging tile
  int lane = threadIdx.x & 31;
  int wave = threadIdx.x >> 5;
  int tile = blockIdx.x * 4 + wave;
  if (tile >= ntiles) return;
  int half = lane >> 4, M = lane & 15;
  int ebase = tile * 16;
  float d = eattr[ebase + M];

  // RBF A-fragment (K=8 padded to 32): gaussian expansion computed in-register
  v8u arbf;
#pragma unroll
  for (int v = 0; v < 8; ++v) {
    int k0 = a_k0(0, v, half);
    if (k0 < 8) {
      float t0 = (d - (float)k0       * (5.0f / 7.0f)) * 1.6f;  // 1/sigma = 8/5
      float t1 = (d - (float)(k0 + 1) * (5.0f / 7.0f)) * 1.6f;
      arbf[v] = packbf(__expf(-t0 * t0), __expf(-t1 * t1));
    } else arbf[v] = 0u;
  }
  int srow[8], drow[8];
#pragma unroll
  for (int r = 0; r < 8; ++r) {
    int e = ebase + r + 8 * half;
    srow[r] = idx_src[e];
    drow[r] = idx_dst[e];
  }

  // base via WMMA, then silu * gathered x_src, stage bf16 into LDS (C->A transpose)
  unsigned short* lus = (unsigned short*)lds[wave];
#pragma unroll
  for (int n = 0; n < 8; ++n) {
    v8f cb = {};
    cb = wmma_bf16(arbf, load_wfrag(WbSwz, n, lane), cb);
    int col = n * 16 + M;
    float bbv = bb[col];
#pragma unroll
    for (int r = 0; r < 8; ++r) {
      float z = cb[r] + bbv;
      float bse = z / (1.0f + __expf(-z));  // SiLU
      float h = bse * x_src[(size_t)srow[r] * HID + col];
      lus[(r + 8 * half) * HID + col] = bf16b(h);
    }
  }

  // reload in A-layout (packed bf16 pairs come back as dwords)
  v8u a[4];
#pragma unroll
  for (int s = 0; s < 4; ++s)
#pragma unroll
    for (int v = 0; v < 8; ++v) {
      int k0 = a_k0(s, v, half);
      a[s][v] = lds[wave][M * 64 + (k0 >> 1)];
    }

  // msg = h @ Wn ; scatter with f32 atomics (L2-resident agg buffer)
#pragma unroll
  for (int n = 0; n < 8; ++n) {
    v8f c = {};
#pragma unroll
    for (int s = 0; s < 4; ++s)
      c = wmma_bf16(a[s], load_wfrag(WnSwz, n * 4 + s, lane), c);
    int col = n * 16 + M;
#pragma unroll
    for (int r = 0; r < 8; ++r)
      atomicAdd(&agg[(size_t)drow[r] * HID + col], c[r]);
  }
  if (M == 0) {
#pragma unroll
    for (int r = 0; r < 8; ++r) atomicAdd(&deg[drow[r]], 1.0f);
  }
}

// ---------------------------------------------------------------------------
// Node update: out = lrelu(x_dst @ Ws + agg/max(deg,1) + b); sum += out
// ---------------------------------------------------------------------------
__global__ void node_update(const float* __restrict__ x_dst, const unsigned* __restrict__ WsSwz,
                            const float* __restrict__ agg, const float* __restrict__ deg,
                            const float* __restrict__ bias,
                            float* __restrict__ out, float* __restrict__ sum, int ntiles) {
  int lane = threadIdx.x & 31;
  int wave = threadIdx.x >> 5;
  int tile = blockIdx.x * 4 + wave;
  if (tile >= ntiles) return;
  int half = lane >> 4, M = lane & 15;
  const float* xr = x_dst + (size_t)(tile * 16 + M) * HID;
  v8u a[4];
#pragma unroll
  for (int s = 0; s < 4; ++s)
#pragma unroll
    for (int v = 0; v < 8; ++v) {
      int k0 = a_k0(s, v, half);
      a[s][v] = packbf(xr[k0], xr[k0 + 1]);
    }
  float dg[8];
#pragma unroll
  for (int r = 0; r < 8; ++r)
    dg[r] = fmaxf(deg[tile * 16 + r + 8 * half], 1.0f);
#pragma unroll
  for (int n = 0; n < 8; ++n) {
    v8f c = {};
#pragma unroll
    for (int s = 0; s < 4; ++s)
      c = wmma_bf16(a[s], load_wfrag(WsSwz, n * 4 + s, lane), c);
    int col = n * 16 + M;
    float bv = bias[col];
#pragma unroll
    for (int r = 0; r < 8; ++r) {
      size_t o = (size_t)(tile * 16 + r + 8 * half) * HID + col;
      float v = c[r] + agg[o] / dg[r] + bv;
      v = v > 0.0f ? v : 0.01f * v;  // leaky_relu
      out[o] = v;
      sum[o] += v;
    }
  }
}

// ---------------------------------------------------------------------------
// Fused edge readout: repr(264) -> relu(@W1)(512) -> @W2(128) = m ;
// w = tanh(m @ Wm) ; atomic segment-sum(w*m) + segment-max(m, ordered-uint)
// ---------------------------------------------------------------------------
__global__ void edge_readout(const float* __restrict__ xl_sum, const float* __restrict__ xp_sum,
                             const float* __restrict__ eattr,
                             const int* __restrict__ esrc, const int* __restrict__ edst,
                             const int* __restrict__ lbatch, const float* __restrict__ Wel,
                             const unsigned* __restrict__ W1Swz, const float* __restrict__ b1,
                             const unsigned* __restrict__ W2Swz, const float* __restrict__ b2,
                             const unsigned* __restrict__ WmSwz, const float* __restrict__ bm,
                             float* __restrict__ out, int ntiles) {
  __shared__ unsigned lds[2][16 * 256];  // per-wave 16x512 bf16 hidden tile
  int lane = threadIdx.x & 31;
  int wave = threadIdx.x >> 5;
  int tile = blockIdx.x * 2 + wave;
  if (tile >= ntiles) return;
  int half = lane >> 4, M = lane & 15;
  int ebase = tile * 16;
  int e = ebase + M;
  int se = esrc[e], de = edst[e];
  float d = eattr[e];
  const float* xlr = xl_sum + (size_t)se * HID;
  const float* xpr = xp_sum + (size_t)de * HID;

  // A fragments of edge_repr (K = 264 padded to 288)
  v8u a1[9];
#pragma unroll
  for (int s = 0; s < 9; ++s)
#pragma unroll
    for (int v = 0; v < 8; ++v) {
      int k0 = a_k0(s, v, half);
      unsigned pw;
      if (k0 < 128)      pw = packbf(xlr[k0], xlr[k0 + 1]);
      else if (k0 < 256) pw = packbf(xpr[k0 - 128], xpr[k0 - 127]);
      else if (k0 < 264) pw = packbf(d * Wel[k0 - 256], d * Wel[k0 - 255]);
      else               pw = 0u;
      a1[s][v] = pw;
    }

  // GEMM1 (264->512) + ReLU, stream hidden into LDS as bf16
  unsigned short* lus = (unsigned short*)lds[wave];
  for (int hn = 0; hn < 32; ++hn) {
    v8f c = {};
#pragma unroll
    for (int s = 0; s < 9; ++s)
      c = wmma_bf16(a1[s], load_wfrag(W1Swz, hn * 9 + s, lane), c);
    int col = hn * 16 + M;
    float bv = b1[col];
#pragma unroll
    for (int r = 0; r < 8; ++r) {
      float h = fmaxf(c[r] + bv, 0.0f);
      lus[(r + 8 * half) * 512 + col] = bf16b(h);
    }
  }

  // GEMM2 (512->128): hold all 16 A fragments, produce m in registers
  v8u a2[16];
#pragma unroll
  for (int s = 0; s < 16; ++s)
#pragma unroll
    for (int v = 0; v < 8; ++v) {
      int k0 = a_k0(s, v, half);
      a2[s][v] = lds[wave][M * 256 + (k0 >> 1)];
    }
  float m[8][8];
#pragma unroll
  for (int n = 0; n < 8; ++n) {
    v8f c = {};
#pragma unroll
    for (int s = 0; s < 16; ++s)
      c = wmma_bf16(a2[s], load_wfrag(W2Swz, n * 16 + s, lane), c);
    int col = n * 16 + M;
    float bv = b2[col];
#pragma unroll
    for (int r = 0; r < 8; ++r) {
      m[n][r] = c[r] + bv;
      lus[(r + 8 * half) * HID + col] = bf16b(m[n][r]);  // reuse LDS for m
    }
  }

  // GEMM3 (128->128) for tanh gate
  v8u a3[4];
#pragma unroll
  for (int s = 0; s < 4; ++s)
#pragma unroll
    for (int v = 0; v < 8; ++v) {
      int k0 = a_k0(s, v, half);
      a3[s][v] = lds[wave][M * 64 + (k0 >> 1)];
    }
  int br[8];
#pragma unroll
  for (int r = 0; r < 8; ++r)
    br[r] = lbatch[esrc[ebase + r + 8 * half]];
  unsigned* outu = (unsigned*)out;
#pragma unroll
  for (int n = 0; n < 8; ++n) {
    v8f c = {};
#pragma unroll
    for (int s = 0; s < 4; ++s)
      c = wmma_bf16(a3[s], load_wfrag(WmSwz, n * 4 + s, lane), c);
    int col = n * 16 + M;
    float bv = bm[col];
#pragma unroll
    for (int r = 0; r < 8; ++r) {
      float mv = m[n][r];
      float wv = tanhf(c[r] + bv);
      atomicAdd(&out[(size_t)br[r] * 256 + col], wv * mv);
      unsigned mu = __float_as_uint(mv);
      unsigned ou = (mu & 0x80000000u) ? ~mu : (mu | 0x80000000u);  // order-preserving
      atomicMax(&outu[(size_t)br[r] * 256 + 128 + col], ou);
    }
  }
}

// Decode ordered-uint maxima; empty segments (sentinel 0 -> NaN) -> 0
__global__ void finalize_max(float* __restrict__ out) {
  int idx = blockIdx.x * blockDim.x + threadIdx.x;
  if (idx >= NGRAPH * 128) return;
  int g = idx >> 7, c = idx & 127;
  unsigned* p = (unsigned*)out + (size_t)g * 256 + 128 + c;
  unsigned o = *p;
  unsigned u = (o & 0x80000000u) ? (o ^ 0x80000000u) : ~o;
  float f = __uint_as_float(u);
  if (!(fabsf(f) <= 3.4e38f)) f = 0.0f;  // NaN / inf -> 0
  ((float*)p)[0] = f;
}

// ---------------------------------------------------------------------------
extern "C" void kernel_launch(void* const* d_in, const int* in_sizes, int n_in,
                              void* d_out, int out_size, void* d_ws, size_t ws_size,
                              hipStream_t stream) {
  (void)in_sizes; (void)n_in; (void)out_size;
  const float* x_lig  = (const float*)d_in[0];
  const float* x_pro  = (const float*)d_in[1];
  const float* eattr  = (const float*)d_in[2];
  const int*   esrc   = (const int*)d_in[3];
  const int*   edst   = (const int*)d_in[4];
  const int*   lbatch = (const int*)d_in[5];
  const float* W_node = (const float*)d_in[6];
  const float* W_el   = (const float*)d_in[7];
  const float* Wb     = (const float*)d_in[8];
  const float* bbp    = (const float*)d_in[9];
  const float* Wn     = (const float*)d_in[10];
  const float* Ws     = (const float*)d_in[11];
  const float* bconv  = (const float*)d_in[12];
  const float* W1     = (const float*)d_in[13];
  const float* b1     = (const float*)d_in[14];
  const float* W2     = (const float*)d_in[15];
  const float* b2     = (const float*)d_in[16];
  const float* Wm     = (const float*)d_in[17];
  const float* bm     = (const float*)d_in[18];

  char* ws = (char*)d_ws;
  size_t off = 0;
  auto alloc = [&](size_t bytes) -> void* {
    off = (off + 255) & ~(size_t)255;
    void* p = ws + off;
    off += bytes;
    return p;
  };
  unsigned* wnodeS = (unsigned*)alloc(32 * 128 * 4);
  unsigned *wbS[6], *wnS[6], *wsS[6];
  for (int i = 0; i < 6; ++i) {
    wbS[i] = (unsigned*)alloc(16 * 128 * 4);
    wnS[i] = (unsigned*)alloc(64 * 128 * 4);
    wsS[i] = (unsigned*)alloc(64 * 128 * 4);
  }
  unsigned* w1S = (unsigned*)alloc(144 * 512 * 4);
  unsigned* w2S = (unsigned*)alloc(256 * 128 * 4);
  unsigned* wmS = (unsigned*)alloc(64 * 128 * 4);
  float* xl0   = (float*)alloc((size_t)NLIG * HID * 4);
  float* xl1   = (float*)alloc((size_t)NLIG * HID * 4);
  float* xlsum = (float*)alloc((size_t)NLIG * HID * 4);
  float* xp0   = (float*)alloc((size_t)NPROT * HID * 4);
  float* xp1   = (float*)alloc((size_t)NPROT * HID * 4);
  float* xpsum = (float*)alloc((size_t)NPROT * HID * 4);
  float* agg   = (float*)alloc((size_t)NPROT * HID * 4);
  float* deg   = (float*)alloc((size_t)NPROT * 4);
  if (off > ws_size) return;

  auto prep = [&](const float* W, unsigned* dst, int Kreal, int Kpad, int N) {
    int total = (Kpad / 2) * N;
    prep_weight<<<(total + 255) / 256, 256, 0, stream>>>(W, dst, Kreal, Kpad, N);
  };
  prep(W_node, wnodeS, 44, 64, 128);
  for (int i = 0; i < 6; ++i) {
    prep(Wb + (size_t)i * 8 * 128,   wbS[i], 8,   32,  128);
    prep(Wn + (size_t)i * 128 * 128, wnS[i], 128, 128, 128);
    prep(Ws + (size_t)i * 128 * 128, wsS[i], 128, 128, 128);
  }
  prep(W1, w1S, 264, 288, 512);
  prep(W2, w2S, 512, 512, 128);
  prep(Wm, wmS, 128, 128, 128);

  hipMemsetAsync(xlsum, 0, (size_t)NLIG * HID * 4, stream);
  hipMemsetAsync(xpsum, 0, (size_t)NPROT * HID * 4, stream);

  const int LT = NLIG / 16, PT = NPROT / 16, ET = NEDGE / 16;
  node_embed<<<(LT + 3) / 4, 128, 0, stream>>>(x_lig, wnodeS, xl0, LT);
  node_embed<<<(PT + 3) / 4, 128, 0, stream>>>(x_pro, wnodeS, xp0, PT);

  float *cl = xl0, *cp = xp0, *tl = xl1, *tp = xp1;
  for (int l = 0; l < 3; ++l) {
    int lp = 2 * l, pl = 2 * l + 1;
    // ligand -> protein
    hipMemsetAsync(agg, 0, (size_t)NPROT * HID * 4, stream);
    hipMemsetAsync(deg, 0, (size_t)NPROT * 4, stream);
    edge_msg<<<(ET + 3) / 4, 128, 0, stream>>>(eattr, esrc, edst, cl, wbS[lp],
                                               bbp + lp * HID, wnS[lp], agg, deg, ET);
    node_update<<<(PT + 3) / 4, 128, 0, stream>>>(cp, wsS[lp], agg, deg,
                                                  bconv + lp * HID, tp, xpsum, PT);
    // protein -> ligand (uses old cp)
    hipMemsetAsync(agg, 0, (size_t)NLIG * HID * 4, stream);
    hipMemsetAsync(deg, 0, (size_t)NLIG * 4, stream);
    edge_msg<<<(ET + 3) / 4, 128, 0, stream>>>(eattr, edst, esrc, cp, wbS[pl],
                                               bbp + pl * HID, wnS[pl], agg, deg, ET);
    node_update<<<(LT + 3) / 4, 128, 0, stream>>>(cl, wsS[pl], agg, deg,
                                                  bconv + pl * HID, tl, xlsum, LT);
    float* t;
    t = cl; cl = tl; tl = t;
    t = cp; cp = tp; tp = t;
  }

  hipMemsetAsync(d_out, 0, (size_t)NGRAPH * 256 * 4, stream);
  edge_readout<<<(ET + 1) / 2, 64, 0, stream>>>(xlsum, xpsum, eattr, esrc, edst,
                                                lbatch, W_el, w1S, b1, w2S, b2,
                                                wmS, bm, (float*)d_out, ET);
  finalize_max<<<(NGRAPH * 128 + 255) / 256, 256, 0, stream>>>((float*)d_out);
}
